// BAKTSimpleX_76544907149500
// MI455X (gfx1250) — matrix operand order repaired and checked by reading, not compiled, not run on hardware
//
#include <hip/hip_runtime.h>
#include <hip/hip_bf16.h>
#include <math.h>

// ---------------------------------------------------------------------------
// Model constants (match reference)
// ---------------------------------------------------------------------------
#define BS   32
#define SEQ  512
#define DIM  256
#define DFF  1024
#define NB   2
#define NH   8
#define DK   32          // DIM / NH

typedef __attribute__((ext_vector_type(16))) __bf16 v16bf;
typedef __attribute__((ext_vector_type(8)))  float  v8f;

static __device__ __forceinline__ v8f wmma_bf16(v16bf a, v16bf b, v8f c) {
  // D(16x16 f32) = A(16x32 bf16) x B(32x16 bf16) + C
  return __builtin_amdgcn_wmma_f32_16x16x32_bf16(
      /*neg_a=*/false, a, /*neg_b=*/false, b,
      /*c_mod=*/(short)0, c, /*reuse_a=*/false, /*reuse_b=*/false);
}

// CDNA5 async direct-to-LDS copy (tracked by ASYNCcnt). 16 bytes per lane.
static __device__ __forceinline__ void async_copy_b128(uint32_t lds_off,
                                                       const __bf16* gptr) {
  asm volatile("global_load_async_to_lds_b128 %0, %1, off"
               :
               : "v"(lds_off), "v"((unsigned long long)(uintptr_t)gptr)
               : "memory");
}
static __device__ __forceinline__ void wait_async0() {
  asm volatile("s_wait_asynccnt 0x0" ::: "memory");
}

// A-matrix 16x32 (MxK), 16-bit: lanes 0-15 -> M=lane, K=0..15;
// lanes 16-31 -> M=lane-16, K=16..31. Source row-major, leading dim ld.
static __device__ __forceinline__ v16bf load_A16(const __bf16* base, int ld, int lane) {
  const int m  = lane & 15;
  const int kb = (lane < 16) ? 0 : 16;
  const __bf16* row = base + (size_t)m * ld + kb;
  v16bf a;
#pragma unroll
  for (int i = 0; i < 8; ++i) a[i] = row[i];
#pragma unroll
  for (int i = 0; i < 8; ++i) a[8 + i] = row[8 + i];
  return a;
}

// B-matrix 32x16 (KxN), 16-bit: lanes 0-15 -> N=lane, K=0..15;
// lanes 16-31 -> N=lane-16, K=16..31. Source stored [N][K] row-major
// (torch Linear weight [out,in], or key rows [seq][dk]).
static __device__ __forceinline__ v16bf load_B16(const __bf16* base, int ld, int lane) {
  const int n  = lane & 15;
  const int kb = (lane < 16) ? 0 : 16;
  const __bf16* row = base + (size_t)n * ld + kb;
  v16bf b;
#pragma unroll
  for (int i = 0; i < 16; ++i) b[i] = row[i];
  return b;
}

// ---------------------------------------------------------------------------
// Elementwise helpers
// ---------------------------------------------------------------------------
__global__ void cvt_f32_bf16_kernel(const float* __restrict__ src,
                                    __bf16* __restrict__ dst, int n) {
  int i = blockIdx.x * blockDim.x + threadIdx.x;
  if (i < n) dst[i] = (__bf16)src[i];
}

__global__ void zero_f32_kernel(float* __restrict__ p, int n) {
  int i = blockIdx.x * blockDim.x + threadIdx.x;
  if (i < n) p[i] = 0.0f;
}

// x = q_embed + pe  (f32 + bf16), y = qa_embed + pe (bf16 only)
__global__ void add_pe_kernel(const float* __restrict__ qe,
                              const float* __restrict__ qa,
                              float* __restrict__ xf,
                              __bf16* __restrict__ xb,
                              __bf16* __restrict__ yb) {
  size_t idx = (size_t)blockIdx.x * blockDim.x + threadIdx.x;
  int dim = (int)(idx % DIM);
  int pos = (int)((idx / DIM) % SEQ);
  float dv  = __expf((float)((dim >> 1) * 2) * (-logf(10000.0f) / (float)DIM));
  float ang = (float)pos * dv;
  float pe  = (dim & 1) ? __cosf(ang) : __sinf(ang);
  float xv = qe[idx] + pe;
  xf[idx] = xv;
  xb[idx] = (__bf16)xv;
  yb[idx] = (__bf16)(qa[idx] + pe);
}

// ---------------------------------------------------------------------------
// Tiled GEMM: C[M,N] = A[M,K](bf16) x W[N,K]^T(bf16) + bias[N], optional relu.
// Block = 256 threads (8 waves) -> 64x128 output tile.
// Wave (4 along M, 2 along N) computes 16x64 via 4 accumulators, so one A
// fragment feeds 4 WMMAs. A/W k-tiles staged into LDS with CDNA5 async
// direct-to-LDS loads (ASYNCcnt) and consumed via ds loads.
// ---------------------------------------------------------------------------
#define GBM 64
#define GBN 128
#define GBK 32

__global__ __launch_bounds__(256)
void gemm_xwT_kernel(const __bf16* __restrict__ A,
                     const __bf16* __restrict__ W,
                     const float* __restrict__ bias,
                     float* __restrict__ Cf,
                     __bf16* __restrict__ Cb,
                     int M, int N, int K, int relu) {
  __shared__ __bf16 As[GBM][GBK];   // 4 KB
  __shared__ __bf16 Ws[GBN][GBK];   // 8 KB

  const int tid  = threadIdx.x;
  const int lane = tid & 31;
  const int wave = tid >> 5;            // 0..7
  const int moff = (wave & 3) * 16;     // wave M sub-tile
  const int noff = (wave >> 2) * 64;    // wave N sub-tile
  const int blockN = blockIdx.x * GBN;
  const int blockM = blockIdx.y * GBM;

  // async staging maps: A = 64 rows x 4 x 16B segs (256), W = 128 x 4 (512)
  const int arow = tid >> 2, aseg = tid & 3;
  const uint32_t ldsA = (uint32_t)(uintptr_t)&As[0][0];
  const uint32_t ldsW = (uint32_t)(uintptr_t)&Ws[0][0];

  v8f acc0 = {}, acc1 = {}, acc2 = {}, acc3 = {};

  for (int k0 = 0; k0 < K; k0 += GBK) {
    // ---- stage A tile (rows blockM..+63, cols k0..k0+31) -----------------
    async_copy_b128(ldsA + (uint32_t)(arow * (GBK * 2) + aseg * 16),
                    A + (size_t)(blockM + arow) * K + k0 + aseg * 8);
    // ---- stage W tile (rows blockN..+127) --------------------------------
#pragma unroll
    for (int j = 0; j < 2; ++j) {
      const int s  = tid * 2 + j;
      const int r  = s >> 2, sg = s & 3;
      async_copy_b128(ldsW + (uint32_t)(r * (GBK * 2) + sg * 16),
                      W + (size_t)(blockN + r) * K + k0 + sg * 8);
    }
    // prefetch next k-tile into GL2 while this one is consumed
    if (k0 + GBK < K) {
      __builtin_prefetch(A + (size_t)(blockM + arow) * K + k0 + GBK, 0, 1);
      __builtin_prefetch(W + (size_t)(blockN + (tid >> 1)) * K + k0 + GBK, 0, 1);
    }
    wait_async0();        // drain this wave's async loads
    __syncthreads();      // all waves' tiles visible

    v16bf a  = load_A16(&As[moff][0], GBK, lane);
    v16bf b0 = load_B16(&Ws[noff +  0][0], GBK, lane);
    v16bf b1 = load_B16(&Ws[noff + 16][0], GBK, lane);
    v16bf b2 = load_B16(&Ws[noff + 32][0], GBK, lane);
    v16bf b3 = load_B16(&Ws[noff + 48][0], GBK, lane);
    acc0 = wmma_bf16(a, b0, acc0);
    acc1 = wmma_bf16(a, b1, acc1);
    acc2 = wmma_bf16(a, b2, acc2);
    acc3 = wmma_bf16(a, b3, acc3);
    __syncthreads();      // protect LDS before next stage
  }

  // ---- epilogue ----------------------------------------------------------
  const int col = lane & 15;
  const int mo  = (lane < 16) ? 0 : 8;
  const int mbase = blockM + moff + mo;
  v8f accs[4] = {acc0, acc1, acc2, acc3};
#pragma unroll
  for (int t = 0; t < 4; ++t) {
    const int n = blockN + noff + t * 16 + col;
    if (n >= N) continue;
    const float bv = bias ? bias[n] : 0.0f;
#pragma unroll
    for (int r = 0; r < 8; ++r) {
      float v = accs[t][r] + bv;
      if (relu) v = fmaxf(v, 0.0f);
      size_t idx = (size_t)(mbase + r) * N + n;
      if (Cf) Cf[idx] = v;
      if (Cb) Cb[idx] = (__bf16)v;
    }
  }
}

// ---------------------------------------------------------------------------
// Attention: q = k (kq_same), strict-causal softmax, zero_pad row 0,
// per-head-averaged score accumulation on last block.
// grid (BS, SEQ/16), 2 waves/block, each wave iterates 4 heads.
// ---------------------------------------------------------------------------
__global__ void attn_kernel(const __bf16* __restrict__ kbuf,  // [BS,SEQ,DIM]
                            const __bf16* __restrict__ vbuf,  // [BS,SEQ,DIM]
                            __bf16* __restrict__ ab,          // [BS,SEQ,DIM] attn out
                            float* __restrict__ scores,       // [BS,SEQ,SEQ] or null
                            int want_scores) {
  const int b    = blockIdx.x;
  const int qt   = blockIdx.y;
  const int q0   = qt * 16;
  const int lane = threadIdx.x & 31;
  const int wave = threadIdx.x >> 5;          // 0..1

  __shared__ float smem[2][16][SEQ];          // 64 KB
  float* myp = &smem[wave][0][0];

  const int nkt  = qt + 1;                    // causal key tiles of 16
  const int kmax = nkt * 16;
  const int kpad = (kmax + 31) & ~31;
  const float scale = 0.17677669529663687f;   // 1/sqrt(32)

  for (int hh = wave; hh < NH; hh += 2) {
    const __bf16* Kh = kbuf + ((size_t)b * SEQ) * DIM + hh * DK;
    const __bf16* Vh = vbuf + ((size_t)b * SEQ) * DIM + hh * DK;

    // ---- scores: S = (Q K^T) * scale, strict-causal mask -----------------
    v16bf aQ = load_A16(Kh + (size_t)q0 * DIM, DIM, lane);  // q == k
    for (int kt = 0; kt < nkt; ++kt) {
      v16bf bK = load_B16(Kh + (size_t)(kt * 16) * DIM, DIM, lane);
      v8f c = {};
      c = wmma_bf16(aQ, bK, c);
      const int col = lane & 15;
      const int mo  = (lane < 16) ? 0 : 8;
      const int j   = kt * 16 + col;
#pragma unroll
      for (int r = 0; r < 8; ++r) {
        int m = mo + r;
        int q = q0 + m;
        myp[m * SEQ + j] = (j < q) ? c[r] * scale : -1e30f;
      }
    }

    // ---- masked softmax (2 lanes per row) --------------------------------
    {
      const int row  = lane & 15;
      const int half = lane >> 4;
      float* prow = myp + row * SEQ;
      const int q = q0 + row;
      float mx = -1e30f;
      for (int j = half; j < kmax; j += 2) mx = fmaxf(mx, prow[j]);
      mx = fmaxf(mx, __shfl_xor(mx, 16, 32));
      float sum = 0.0f;
      for (int j = half; j < kmax; j += 2) {
        float e = __expf(prow[j] - mx);
        prow[j] = e;
        sum += e;
      }
      sum += __shfl_xor(sum, 16, 32);
      const float inv = (q == 0) ? 0.0f : 1.0f / sum;   // zero_pad first row
      for (int j = half; j < kmax; j += 2) {
        float p = prow[j] * inv;
        prow[j] = p;
        if (want_scores)
          atomicAdd(&scores[((size_t)b * SEQ + q) * SEQ + j], p * (1.0f / NH));
      }
      for (int j = kmax + half; j < kpad; j += 2) prow[j] = 0.0f;
    }

    // ---- O = P x V (two 16-wide dim halves, K-chunks of 32 keys) ---------
    v8f o0 = {}, o1 = {};
    {
      const int m   = lane & 15;
      const int kbA = (lane < 16) ? 0 : 16;
      const int nn  = lane & 15;
      const int kbB = (lane < 16) ? 0 : 16;
      for (int kc = 0; kc < kpad; kc += 32) {
        v16bf aP;
        const float* pr = myp + m * SEQ + kc + kbA;
#pragma unroll
        for (int i = 0; i < 8; ++i) aP[i]     = (__bf16)pr[i];
#pragma unroll
        for (int i = 0; i < 8; ++i) aP[8 + i] = (__bf16)pr[8 + i];
        v16bf b0, b1;
#pragma unroll
        for (int i = 0; i < 16; ++i) {
          const __bf16* vrow = Vh + (size_t)(kc + kbB + i) * DIM;
          b0[i] = vrow[nn];
          b1[i] = vrow[16 + nn];
        }
        o0 = wmma_bf16(aP, b0, o0);
        o1 = wmma_bf16(aP, b1, o1);
      }
    }

    // ---- store attn output [b, q, h*dk + dim] ----------------------------
    {
      const int col = lane & 15;
      const int mo  = (lane < 16) ? 0 : 8;
      __bf16* dst = ab + ((size_t)b * SEQ + q0) * DIM + hh * DK;
#pragma unroll
      for (int r = 0; r < 8; ++r) {
        dst[(size_t)(mo + r) * DIM + col]      = (__bf16)o0[r];
        dst[(size_t)(mo + r) * DIM + 16 + col] = (__bf16)o1[r];
      }
    }
  }
}

// ---------------------------------------------------------------------------
// Fused residual + LayerNorm (+ bf16 requant). One row per block (d = 256).
// ---------------------------------------------------------------------------
__global__ void add_ln_kernel(const float* __restrict__ a,
                              const float* __restrict__ bsrc,
                              const float* __restrict__ g,
                              const float* __restrict__ be,
                              float* __restrict__ outf,
                              __bf16* __restrict__ outb) {
  const int row = blockIdx.x;
  const int t   = threadIdx.x;     // blockDim.x == DIM
  __shared__ float red[DIM];
  const size_t base = (size_t)row * DIM;

  float z = a[base + t] + bsrc[base + t];
  red[t] = z;
  __syncthreads();
  for (int off = DIM / 2; off > 0; off >>= 1) {
    if (t < off) red[t] += red[t + off];
    __syncthreads();
  }
  const float mean = red[0] * (1.0f / DIM);
  __syncthreads();
  const float dz = z - mean;
  red[t] = dz * dz;
  __syncthreads();
  for (int off = DIM / 2; off > 0; off >>= 1) {
    if (t < off) red[t] += red[t + off];
    __syncthreads();
  }
  const float var = red[0] * (1.0f / DIM);
  const float y = dz * rsqrtf(var + 1e-5f) * g[t] + be[t];
  outf[base + t] = y;
  if (outb) outb[base + t] = (__bf16)y;
}

// ---------------------------------------------------------------------------
// Host-side orchestration
// ---------------------------------------------------------------------------
extern "C" void kernel_launch(void* const* d_in, const int* in_sizes, int n_in,
                              void* d_out, int out_size, void* d_ws, size_t ws_size,
                              hipStream_t stream) {
  (void)in_sizes; (void)n_in; (void)out_size; (void)ws_size;

  const float* qe    = (const float*)d_in[0];
  const float* qa    = (const float*)d_in[1];
  const float* Wk    = (const float*)d_in[2];
  const float* bk    = (const float*)d_in[3];
  const float* Wv    = (const float*)d_in[4];
  const float* bv    = (const float*)d_in[5];
  const float* Wo    = (const float*)d_in[6];
  const float* bo    = (const float*)d_in[7];
  const float* ln1g  = (const float*)d_in[8];
  const float* ln1b  = (const float*)d_in[9];
  const float* W1    = (const float*)d_in[10];
  const float* b1    = (const float*)d_in[11];
  const float* W2    = (const float*)d_in[12];
  const float* b2    = (const float*)d_in[13];
  const float* ln2g  = (const float*)d_in[14];
  const float* ln2b  = (const float*)d_in[15];

  float* out_x      = (float*)d_out;                       // [BS,SEQ,DIM]
  float* out_scores = out_x + (size_t)BS * SEQ * DIM;      // [BS,SEQ,SEQ]

  const size_t M   = (size_t)BS * SEQ;                     // 16384 rows
  const size_t ACT = M * DIM;                              // activation elems
  const size_t FF  = M * DFF;

  // ---- workspace carve ---------------------------------------------------
  char* p = (char*)d_ws;
  auto carve = [&](size_t bytes) {
    char* r = p;
    p += (bytes + 255) & ~(size_t)255;
    return (void*)r;
  };
  float*  xf  = (float*) carve(ACT * 4);      // x stream f32
  __bf16* xb  = (__bf16*)carve(ACT * 2);      // x stream bf16
  __bf16* yb  = (__bf16*)carve(ACT * 2);      // y stream bf16
  __bf16* kb_ = (__bf16*)carve(ACT * 2);      // K (=Q) projection
  __bf16* vb_ = (__bf16*)carve(ACT * 2);      // V projection
  __bf16* abf = (__bf16*)carve(ACT * 2);      // attention output
  float*  of  = (float*) carve(ACT * 4);      // O-projection out
  __bf16* f1b = (__bf16*)carve(FF  * 2);      // FFN hidden (relu'd)
  float*  f2f = (float*) carve(ACT * 4);      // FFN out
  __bf16* Wkb = (__bf16*)carve((size_t)NB * DIM * DIM * 2);
  __bf16* Wvb = (__bf16*)carve((size_t)NB * DIM * DIM * 2);
  __bf16* Wob = (__bf16*)carve((size_t)NB * DIM * DIM * 2);
  __bf16* W1b = (__bf16*)carve((size_t)NB * DFF * DIM * 2);
  __bf16* W2b = (__bf16*)carve((size_t)NB * DIM * DFF * 2);

  // ---- weight f32 -> bf16 ------------------------------------------------
  {
    const int nw1 = NB * DIM * DIM;
    const int nw2 = NB * DFF * DIM;
    cvt_f32_bf16_kernel<<<(nw1 + 255) / 256, 256, 0, stream>>>(Wk, Wkb, nw1);
    cvt_f32_bf16_kernel<<<(nw1 + 255) / 256, 256, 0, stream>>>(Wv, Wvb, nw1);
    cvt_f32_bf16_kernel<<<(nw1 + 255) / 256, 256, 0, stream>>>(Wo, Wob, nw1);
    cvt_f32_bf16_kernel<<<(nw2 + 255) / 256, 256, 0, stream>>>(W1, W1b, nw2);
    cvt_f32_bf16_kernel<<<(nw2 + 255) / 256, 256, 0, stream>>>(W2, W2b, nw2);
  }

  // ---- positional embedding add ------------------------------------------
  add_pe_kernel<<<(unsigned)(ACT / 256), 256, 0, stream>>>(qe, qa, xf, xb, yb);

  // ---- transformer blocks ------------------------------------------------
  for (int i = 0; i < NB; ++i) {
    const int last = (i == NB - 1);

    // K (=Q) projection: kb_ = xb @ Wk[i]^T + bk[i]
    gemm_xwT_kernel<<<dim3(DIM / GBN, (unsigned)(M / GBM)), 256, 0, stream>>>(
        xb, Wkb + (size_t)i * DIM * DIM, bk + (size_t)i * DIM,
        nullptr, kb_, (int)M, DIM, DIM, 0);

    // V projection: vb_ = yb @ Wv[i]^T + bv[i]
    gemm_xwT_kernel<<<dim3(DIM / GBN, (unsigned)(M / GBM)), 256, 0, stream>>>(
        yb, Wvb + (size_t)i * DIM * DIM, bv + (size_t)i * DIM,
        nullptr, vb_, (int)M, DIM, DIM, 0);

    // scores output zeroed before accumulation on last block
    if (last) {
      const int ns = BS * SEQ * SEQ;
      zero_f32_kernel<<<(ns + 255) / 256, 256, 0, stream>>>(out_scores, ns);
    }

    // attention
    attn_kernel<<<dim3(BS, SEQ / 16), 64, 0, stream>>>(
        kb_, vb_, abf, last ? out_scores : nullptr, last);

    // O projection: of = abf @ Wo[i]^T + bo[i]  (f32 for residual)
    gemm_xwT_kernel<<<dim3(DIM / GBN, (unsigned)(M / GBM)), 256, 0, stream>>>(
        abf, Wob + (size_t)i * DIM * DIM, bo + (size_t)i * DIM,
        of, nullptr, (int)M, DIM, DIM, 0);

    // x = LN1(x + attn)
    add_ln_kernel<<<(unsigned)M, DIM, 0, stream>>>(
        xf, of, ln1g + (size_t)i * DIM, ln1b + (size_t)i * DIM, xf, xb);

    // FFN hidden: f1b = relu(xb @ W1[i]^T + b1[i])
    gemm_xwT_kernel<<<dim3(DFF / GBN, (unsigned)(M / GBM)), 256, 0, stream>>>(
        xb, W1b + (size_t)i * DFF * DIM, b1 + (size_t)i * DFF,
        nullptr, f1b, (int)M, DFF, DIM, 1);

    // FFN out: f2f = f1b @ W2[i]^T + b2[i]
    gemm_xwT_kernel<<<dim3(DIM / GBN, (unsigned)(M / GBM)), 256, 0, stream>>>(
        f1b, W2b + (size_t)i * DIM * DFF, b2 + (size_t)i * DIM,
        f2f, nullptr, (int)M, DIM, DFF, 0);

    // x = LN2(x + ff)  (last block writes f32 straight to d_out)
    add_ln_kernel<<<(unsigned)M, DIM, 0, stream>>>(
        xf, f2f, ln2g + (size_t)i * DIM, ln2b + (size_t)i * DIM,
        last ? out_x : xf, last ? nullptr : xb);
  }
}
